// TGAT_Layer_89558658056627
// MI455X (gfx1250) — compile-verified, matrix operand order checked by reading, not run on hardware
//
#include <hip/hip_runtime.h>
#include <hip/hip_bf16.h>

// TGAT layer for MI455X (gfx1250, wave32, WMMA bf16 + async global->LDS).
// N=8192 nodes, D=32 neighbors, F=256, H=8 heads (HD=32), T=64, R=64.

typedef __bf16 bf16;
typedef __attribute__((ext_vector_type(16))) __bf16 v16bf;
typedef __attribute__((ext_vector_type(8)))  float  v8f;
typedef __attribute__((ext_vector_type(4)))  unsigned int uv4;

#define WMMA_BF16(a, b, c) \
  __builtin_amdgcn_wmma_f32_16x16x32_bf16(false, (a), false, (b), (short)0, (c), false, false)

__device__ __forceinline__ bf16 f2bf(float f) {
  unsigned u = __builtin_bit_cast(unsigned, f);
  u += 0x7FFFu + ((u >> 16) & 1u);             // round-to-nearest-even
  unsigned short h = (unsigned short)(u >> 16);
  return __builtin_bit_cast(bf16, h);
}
__device__ __forceinline__ float bf2f(bf16 b) {
  unsigned short h = __builtin_bit_cast(unsigned short, b);
  unsigned u = ((unsigned)h) << 16;
  return __builtin_bit_cast(float, u);
}
__device__ __forceinline__ v8f vzero8() {
  v8f z = {0.f, 0.f, 0.f, 0.f, 0.f, 0.f, 0.f, 0.f};
  return z;
}

// Async DMA: 16 bytes/lane global -> LDS, tracked by ASYNCcnt (no VGPR data).
// ISA 15.18.3 op 98: the instruction offset is added to both the LDS and global address.
__device__ __forceinline__ void async_copy_b128(unsigned lds_off, const void* gaddr) {
  asm volatile("global_load_async_to_lds_b128 %0, %1, off"
               :: "v"(lds_off), "v"(gaddr) : "memory");
}
__device__ __forceinline__ void async_copy_b128_o16(unsigned lds_off, const void* gaddr) {
  asm volatile("global_load_async_to_lds_b128 %0, %1, off offset:16"
               :: "v"(lds_off), "v"(gaddr) : "memory");
}
__device__ __forceinline__ void wait_async0() {
#if __has_builtin(__builtin_amdgcn_s_wait_asynccnt)
  __builtin_amdgcn_s_wait_asynccnt(0);
#else
  asm volatile("s_wait_asynccnt 0x0" ::: "memory");
#endif
}

// Load one 16x32 bf16 WMMA fragment from a row-major [rows][ld] matrix.
// lane 0-15 : row row0+lane,    K halves k0+0..7  and k0+16..23
// lane 16-31: row row0+lane-16, K halves k0+8..15 and k0+24..31
__device__ __forceinline__ v16bf load_frag(const bf16* base, int ld, int row0, int k0, int lane) {
  int r  = row0 + (lane & 15);
  int kk = k0 + ((lane >> 4) << 3);
  const bf16* p = base + (size_t)r * ld + kk;
  union { v16bf v; uv4 q[2]; } t;
  t.q[0] = *(const uv4*)(p);
  t.q[1] = *(const uv4*)(p + 16);
  return t.v;
}

// ---------------------------------------------------------------- weights
// Wkqv  [768][256]  <- kqv_w  [256][768]^T
// Wmod  [512][128]  : out o<256 -> col o      of [t_kqv_w; e_kqv_w] (k part)
//                     out o>=256-> col 512+o-256 (v part); in i<64 -> t, else e
// Wlin1 [256][512]  <- lin1_w [512][256]^T
// Wlin2 [256][256]  <- lin2_w [256][256]^T
__global__ void prep_weights_kernel(const float* __restrict__ kqv_w,
                                    const float* __restrict__ t_kqv_w,
                                    const float* __restrict__ e_kqv_w,
                                    const float* __restrict__ lin1_w,
                                    const float* __restrict__ lin2_w,
                                    bf16* __restrict__ Wkqv, bf16* __restrict__ Wmod,
                                    bf16* __restrict__ Wlin1, bf16* __restrict__ Wlin2) {
  int idx = blockIdx.x * blockDim.x + threadIdx.x;
  if (idx < 196608) {                       // 768*256
    int o = idx >> 8, i = idx & 255;
    Wkqv[idx] = f2bf(kqv_w[i * 768 + o]);
    return;
  }
  idx -= 196608;
  if (idx < 65536) {                        // 512*128
    int o = idx >> 7, i = idx & 127;
    int col = (o < 256) ? o : (256 + o);    // v part lives at 512+(o-256)
    float v = (i < 64) ? t_kqv_w[i * 768 + col] : e_kqv_w[(i - 64) * 768 + col];
    Wmod[idx] = f2bf(v);
    return;
  }
  idx -= 65536;
  if (idx < 131072) {                       // 256*512
    int o = idx >> 9, i = idx & 511;
    Wlin1[idx] = f2bf(lin1_w[i * 256 + o]);
    return;
  }
  idx -= 131072;
  if (idx < 65536) {                        // 256*256
    int o = idx >> 8, i = idx & 255;
    Wlin2[idx] = f2bf(lin2_w[i * 256 + o]);
  }
}

// t_q0[c] = t2v(start) @ t_kqv_w[:, 256:512]
__global__ void tq0_kernel(const float* __restrict__ t_kqv_w,
                           const float* __restrict__ w0p, const float* __restrict__ b0p,
                           const float* __restrict__ t2v_w, const float* __restrict__ t2v_b,
                           const int* __restrict__ start_p, float* __restrict__ tq0) {
  int c = threadIdx.x;
  float st = (float)start_p[0];
  float acc = 0.f;
  for (int t = 0; t < 64; ++t) {
    float te = (t == 0) ? (st * w0p[0] + b0p[0]) : __sinf(st * t2v_w[t - 1] + t2v_b[t - 1]);
    acc += te * t_kqv_w[t * 768 + 256 + c];
  }
  tq0[c] = acc;
}

// LayerNorm: one wave per row (256 feats, 8 per lane). Writes optional fp32 copy
// and a bf16 copy at cc[row*512 + bf_off + c].
__global__ void ln_kernel(const float* __restrict__ x, const float* __restrict__ g,
                          const float* __restrict__ b, float* __restrict__ out_f32,
                          bf16* __restrict__ cc, int bf_off) {
  int w = threadIdx.x >> 5, lane = threadIdx.x & 31;
  int row = blockIdx.x * 8 + w;
  const float* xr = x + (size_t)row * 256;
  float v[8];
  float s = 0.f;
#pragma unroll
  for (int k = 0; k < 8; ++k) { v[k] = xr[k * 32 + lane]; s += v[k]; }
  for (int off = 16; off > 0; off >>= 1) s += __shfl_xor(s, off, 32);
  float mu = s * (1.f / 256.f);
  float vs = 0.f;
#pragma unroll
  for (int k = 0; k < 8; ++k) { float d = v[k] - mu; vs += d * d; }
  for (int off = 16; off > 0; off >>= 1) vs += __shfl_xor(vs, off, 32);
  float rinv = rsqrtf(vs * (1.f / 256.f) + 1e-5f);
#pragma unroll
  for (int k = 0; k < 8; ++k) {
    int c = k * 32 + lane;
    float o = (v[k] - mu) * rinv * g[c] + b[c];
    if (out_f32) out_f32[(size_t)row * 256 + c] = o;
    cc[(size_t)row * 512 + bf_off + c] = f2bf(o);
  }
}

// kqv = xn(bf16, ld 512) @ Wkqv^T  -> fp32 [N,768]
__global__ void __launch_bounds__(256)
gemm_kqv_kernel(const bf16* __restrict__ A, const bf16* __restrict__ B,
                float* __restrict__ C) {
  const int lane = threadIdx.x & 31, w = threadIdx.x >> 5;
  const int row0 = blockIdx.y * 32 + (w >> 2) * 16;
  const int col0 = blockIdx.x * 256 + (w & 3) * 64;
  v8f acc[4] = {vzero8(), vzero8(), vzero8(), vzero8()};
  for (int k0 = 0; k0 < 256; k0 += 32) {
    v16bf a = load_frag(A, 512, row0, k0, lane);
#pragma unroll
    for (int nt = 0; nt < 4; ++nt) {
      v16bf b = load_frag(B, 256, col0 + nt * 16, k0, lane);
      acc[nt] = WMMA_BF16(a, b, acc[nt]);
    }
  }
  int crow = row0 + ((lane >> 4) << 3);
  int ccol = col0 + (lane & 15);
#pragma unroll
  for (int nt = 0; nt < 4; ++nt)
#pragma unroll
    for (int j = 0; j < 8; ++j)
      C[(size_t)(crow + j) * 768 + ccol + nt * 16] = acc[nt][j];
}

// Fused per-node kernel: async rels DMA + Time2Vec + edge GEMM (WMMA) + gather + attention.
__global__ void __launch_bounds__(256)
edge_attn_kernel(const int* __restrict__ neighbors, const float* __restrict__ times,
                 const float* __restrict__ rels, const bf16* __restrict__ Wmod,
                 const float* __restrict__ kqv, const float* __restrict__ tq0,
                 const float* __restrict__ t2v_w, const float* __restrict__ t2v_b,
                 const float* __restrict__ w0p, const float* __restrict__ b0p,
                 const int* __restrict__ start_p, const int* __restrict__ end_p,
                 const float* __restrict__ xn, float* __restrict__ h2) {
  __shared__ float rels_lds[32 * 64];  // 8 KB, filled by async DMA
  __shared__ bf16 A_lds[32 * 128];     // [edge][te(64) | rels(64)]
  __shared__ bf16 kv_lds[32 * 512];    // [edge][k(256) | v(256)]
  __shared__ float q_lds[256];
  __shared__ float times_lds[32];
  __shared__ int nbr_lds[32];

  const int n = blockIdx.x, tid = threadIdx.x;
  const int lane = tid & 31, w = tid >> 5;
  const float w0 = w0p[0], b0 = b0p[0];
  const float startf = (float)start_p[0], endf = (float)end_p[0];
  const int e = tid >> 3;              // 8 threads per edge
  const int sub = tid & 7;             // 8-float sub-chunk within the edge

  // kick off the rels tile DMA (32x64 f32 = 8KB): 32 bytes per thread,
  // overlapped with the Time2Vec transcendental work below.
  {
    unsigned lds_off = (unsigned)(size_t)(void*)rels_lds + (unsigned)tid * 32u;
    const float* gsrc = rels + (size_t)n * 2048 + (size_t)tid * 8;
    async_copy_b128(lds_off, gsrc);
    async_copy_b128_o16(lds_off, gsrc);
  }

  if (tid < 32) {
    nbr_lds[tid] = neighbors[n * 32 + tid];
    times_lds[tid] = times[n * 32 + tid];
  }
  q_lds[tid] = kqv[(size_t)n * 768 + 256 + tid] + tq0[tid];

  {  // Time2Vec features -> A cols 0..63 (overlaps the async DMA)
    float tm = times[n * 32 + e];
    int cb = sub * 8;
#pragma unroll
    for (int j = 0; j < 8; ++j) {
      int c = cb + j;
      float v = (c == 0) ? (tm * w0 + b0) : __sinf(tm * t2v_w[c - 1] + t2v_b[c - 1]);
      A_lds[e * 128 + c] = f2bf(v);
    }
  }

  wait_async0();
  __syncthreads();

  {  // rels -> bf16 A cols 64..127
    int rb = sub * 8;
#pragma unroll
    for (int j = 0; j < 8; ++j)
      A_lds[e * 128 + 64 + rb + j] = f2bf(rels_lds[e * 64 + rb + j]);
  }
  __syncthreads();

  // GEMM [32 x 128] @ [128 x 512]: wave w owns cols w*64 .. w*64+63
  v8f acc[2][4];
#pragma unroll
  for (int mt = 0; mt < 2; ++mt)
#pragma unroll
    for (int nt = 0; nt < 4; ++nt) acc[mt][nt] = vzero8();
  for (int k0 = 0; k0 < 128; k0 += 32) {
    v16bf a0 = load_frag(A_lds, 128, 0, k0, lane);
    v16bf a1 = load_frag(A_lds, 128, 16, k0, lane);
#pragma unroll
    for (int nt = 0; nt < 4; ++nt) {
      v16bf b = load_frag(Wmod, 128, w * 64 + nt * 16, k0, lane);
      acc[0][nt] = WMMA_BF16(a0, b, acc[0][nt]);
      acc[1][nt] = WMMA_BF16(a1, b, acc[1][nt]);
    }
  }
  // epilogue: add gathered k_/v_ rows, park k/v in LDS (bf16)
  const int colb = w * 64 + (lane & 15);
  const int rowo = (lane >> 4) << 3;
#pragma unroll
  for (int mt = 0; mt < 2; ++mt)
#pragma unroll
    for (int nt = 0; nt < 4; ++nt) {
      int col = colb + nt * 16;
      int src = (col < 256) ? col : (256 + col);  // k at [0,256), v at [512,768)
#pragma unroll
      for (int j = 0; j < 8; ++j) {
        int ee = mt * 16 + rowo + j;
        float gval = kqv[(size_t)nbr_lds[ee] * 768 + src];
        kv_lds[ee * 512 + col] = f2bf(acc[mt][nt][j] + gval);
      }
    }
  __syncthreads();

  // attention: wave = head, lane = edge (and lane = dim for output)
  {
    const int h = w, el = lane;
    float tm = times_lds[el];
    bool valid = (tm >= startf) && (tm < endf);
    float lg = 0.f;
    const bf16* krow = &kv_lds[el * 512 + h * 32];
#pragma unroll
    for (int d = 0; d < 32; ++d) lg += q_lds[h * 32 + d] * bf2f(krow[d]);
    lg *= 0.17677669529663687f;  // 1/sqrt(32)
    if (!valid) lg = -1e9f;
    float m = lg;
    for (int off = 16; off > 0; off >>= 1) m = fmaxf(m, __shfl_xor(m, off, 32));
    float p = valid ? __expf(lg - m) : 0.f;
    float s = p;
    for (int off = 16; off > 0; off >>= 1) s += __shfl_xor(s, off, 32);
    float at = (s > 0.f) ? (p / s) : 0.f;
    bool has = __any(valid);
    float o = 0.f;
#pragma unroll
    for (int e2 = 0; e2 < 32; ++e2) {
      float a = __shfl(at, e2, 32);
      o += a * bf2f(kv_lds[e2 * 512 + 256 + h * 32 + lane]);
    }
    int c = h * 32 + lane;
    float outv = has ? o : 0.f;
    h2[(size_t)n * 256 + c] = outv + xn[(size_t)n * 256 + c];
  }
}

// act1 = relu(cc[N,512] @ Wlin1^T + b1) -> bf16 [N,256]
__global__ void __launch_bounds__(256)
gemm_lin1_kernel(const bf16* __restrict__ A, const bf16* __restrict__ B,
                 const float* __restrict__ bias, bf16* __restrict__ Cb) {
  const int lane = threadIdx.x & 31, w = threadIdx.x >> 5;
  const int row0 = blockIdx.y * 32 + (w >> 2) * 16;
  const int col0 = (w & 3) * 64;
  v8f acc[4] = {vzero8(), vzero8(), vzero8(), vzero8()};
  for (int k0 = 0; k0 < 512; k0 += 32) {
    v16bf a = load_frag(A, 512, row0, k0, lane);
#pragma unroll
    for (int nt = 0; nt < 4; ++nt) {
      v16bf b = load_frag(B, 512, col0 + nt * 16, k0, lane);
      acc[nt] = WMMA_BF16(a, b, acc[nt]);
    }
  }
  int crow = row0 + ((lane >> 4) << 3);
  int ccol = col0 + (lane & 15);
#pragma unroll
  for (int nt = 0; nt < 4; ++nt) {
    int col = ccol + nt * 16;
    float bsv = bias[col];
#pragma unroll
    for (int j = 0; j < 8; ++j) {
      float r = fmaxf(acc[nt][j] + bsv, 0.f);
      Cb[(size_t)(crow + j) * 256 + col] = f2bf(r);
    }
  }
}

// out = act1[N,256] @ Wlin2^T + b2 + h2 -> fp32 d_out
__global__ void __launch_bounds__(256)
gemm_lin2_kernel(const bf16* __restrict__ A, const bf16* __restrict__ B,
                 const float* __restrict__ bias, const float* __restrict__ h2,
                 float* __restrict__ C) {
  const int lane = threadIdx.x & 31, w = threadIdx.x >> 5;
  const int row0 = blockIdx.y * 32 + (w >> 2) * 16;
  const int col0 = (w & 3) * 64;
  v8f acc[4] = {vzero8(), vzero8(), vzero8(), vzero8()};
  for (int k0 = 0; k0 < 256; k0 += 32) {
    v16bf a = load_frag(A, 256, row0, k0, lane);
#pragma unroll
    for (int nt = 0; nt < 4; ++nt) {
      v16bf b = load_frag(B, 256, col0 + nt * 16, k0, lane);
      acc[nt] = WMMA_BF16(a, b, acc[nt]);
    }
  }
  int crow = row0 + ((lane >> 4) << 3);
  int ccol = col0 + (lane & 15);
#pragma unroll
  for (int nt = 0; nt < 4; ++nt) {
    int col = ccol + nt * 16;
    float bsv = bias[col];
#pragma unroll
    for (int j = 0; j < 8; ++j) {
      size_t ix = (size_t)(crow + j) * 256 + col;
      C[ix] = acc[nt][j] + bsv + h2[ix];
    }
  }
}

extern "C" void kernel_launch(void* const* d_in, const int* in_sizes, int n_in,
                              void* d_out, int out_size, void* d_ws, size_t ws_size,
                              hipStream_t stream) {
  (void)in_sizes; (void)n_in; (void)out_size; (void)ws_size;
  const float* x       = (const float*)d_in[0];
  const int*   nbrs    = (const int*)d_in[1];
  const float* times   = (const float*)d_in[2];
  const float* rels    = (const float*)d_in[3];
  const float* kqv_w   = (const float*)d_in[4];
  const float* t_kqv_w = (const float*)d_in[5];
  const float* e_kqv_w = (const float*)d_in[6];
  const float* t2v_w0  = (const float*)d_in[7];
  const float* t2v_b0  = (const float*)d_in[8];
  const float* t2v_w   = (const float*)d_in[9];
  const float* t2v_b   = (const float*)d_in[10];
  const float* ln1_g   = (const float*)d_in[11];
  const float* ln1_b   = (const float*)d_in[12];
  const float* ln2_g   = (const float*)d_in[13];
  const float* ln2_b   = (const float*)d_in[14];
  const float* lin1_w  = (const float*)d_in[15];
  const float* lin1_b  = (const float*)d_in[16];
  const float* lin2_w  = (const float*)d_in[17];
  const float* lin2_b  = (const float*)d_in[18];
  const int*   start_p = (const int*)d_in[19];
  const int*   end_p   = (const int*)d_in[20];
  float* out = (float*)d_out;

  char* ws = (char*)d_ws;
  const size_t MB = 1u << 20;
  bf16*  Wkqv  = (bf16*)(ws + 0);          // 768*256*2  = 384 KB
  bf16*  Wmod  = (bf16*)(ws + 393216);     // 512*128*2  = 128 KB
  bf16*  Wlin1 = (bf16*)(ws + 524288);     // 256*512*2  = 256 KB
  bf16*  Wlin2 = (bf16*)(ws + 786432);     // 256*256*2  = 128 KB
  float* tq0   = (float*)(ws + 917504);    // 1 KB
  bf16*  cc    = (bf16*)(ws + 1 * MB);     // [N,512] bf16 = 8 MB  (xn | hn)
  float* xn    = (float*)(ws + 9 * MB);    // [N,256] f32  = 8 MB
  float* kqv   = (float*)(ws + 17 * MB);   // [N,768] f32  = 24 MB
  float* h2    = (float*)(ws + 41 * MB);   // [N,256] f32  = 8 MB
  bf16*  act1  = (bf16*)(ws + 49 * MB);    // [N,256] bf16 = 4 MB

  prep_weights_kernel<<<dim3(1792), dim3(256), 0, stream>>>(
      kqv_w, t_kqv_w, e_kqv_w, lin1_w, lin2_w, Wkqv, Wmod, Wlin1, Wlin2);
  tq0_kernel<<<dim3(1), dim3(256), 0, stream>>>(
      t_kqv_w, t2v_w0, t2v_b0, t2v_w, t2v_b, start_p, tq0);
  ln_kernel<<<dim3(1024), dim3(256), 0, stream>>>(x, ln1_g, ln1_b, xn, cc, 0);
  gemm_kqv_kernel<<<dim3(3, 256), dim3(256), 0, stream>>>(cc, Wkqv, kqv);
  edge_attn_kernel<<<dim3(8192), dim3(256), 0, stream>>>(
      nbrs, times, rels, Wmod, kqv, tq0, t2v_w, t2v_b, t2v_w0, t2v_b0,
      start_p, end_p, xn, h2);
  ln_kernel<<<dim3(1024), dim3(256), 0, stream>>>(h2, ln2_g, ln2_b, nullptr, cc, 256);
  gemm_lin1_kernel<<<dim3(1, 256), dim3(256), 0, stream>>>(cc, Wlin1, lin1_b, act1);
  gemm_lin2_kernel<<<dim3(1, 256), dim3(256), 0, stream>>>(act1, Wlin2, lin2_b, h2, out);
}
